// CTransformerBlock_19318762897742
// MI455X (gfx1250) — matrix operand order repaired and checked by reading, not compiled, not run on hardware
//
#include <hip/hip_runtime.h>
#include <hip/hip_bf16.h>

#define BB  4
#define NN  2048
#define KNB 36
#define DM  128
#define DP  32

typedef __attribute__((ext_vector_type(16))) _Float16 v16h;
typedef __attribute__((ext_vector_type(8)))  _Float16 h8;
typedef __attribute__((ext_vector_type(8)))  float    v8f;

// ---------------------------------------------------------------------------
// WMMA helpers (CDNA5 16x16x32 f16 -> f32)
// ---------------------------------------------------------------------------
__device__ __forceinline__ v8f wmma32(v16h a, v16h b, v8f c) {
  // (neg_a, A, neg_b, B, c_mod, C, reuse_a, reuse_b)
  return __builtin_amdgcn_wmma_f32_16x16x32_f16(false, a, false, b, (short)0, c, false, false);
}

// B fragment: column N = nt*16+lm, K groups {kc*32+hi*8 .. +7} and {kc*32+16+hi*8 .. +7}
// Weights stored row-major [out][in] == exactly w^T as the GEMM needs.
__device__ __forceinline__ v16h load_b_frag(const _Float16* W, int nt, int kc, int lm, int hi) {
  const h8* p0 = (const h8*)(W + (nt * 16 + lm) * DM + kc * 32 + hi * 8);
  const h8* p1 = (const h8*)(W + (nt * 16 + lm) * DM + kc * 32 + 16 + hi * 8);
  h8 x0 = *p0, x1 = *p1;
  v16h b;
#pragma unroll
  for (int t = 0; t < 8; ++t) { b[t] = x0[t]; b[t + 8] = x1[t]; }
  return b;
}

// A fragment from a row-major 16xDM f16 buffer:
// row M = lm, K groups {kc*32+hi*8 .. +7} and {kc*32+16+hi*8 .. +7}
__device__ __forceinline__ v16h load_a_frag(const _Float16* Abuf, int kc, int lm, int hi) {
  const h8* p0 = (const h8*)(Abuf + lm * DM + kc * 32 + hi * 8);
  const h8* p1 = (const h8*)(Abuf + lm * DM + kc * 32 + 16 + hi * 8);
  h8 x0 = *p0, x1 = *p1;
  v16h a;
#pragma unroll
  for (int t = 0; t < 8; ++t) { a[t] = x0[t]; a[t + 8] = x1[t]; }
  return a;
}

// ---------------------------------------------------------------------------
// Kernel 1: exact KNN (top-36 smallest squared distances, index-tie-break)
// one block per query point
// ---------------------------------------------------------------------------
__global__ __launch_bounds__(256) void knn_kernel(const float* __restrict__ xyz,
                                                  int* __restrict__ knn) {
  __shared__ float sx[NN * 3];
  __shared__ float sd[NN];
  __shared__ float rv[256];
  __shared__ int   ri[256];
  const int tid = threadIdx.x;
  const int p = blockIdx.x;
  const int b = p / NN, n = p % NN;
  for (int i = tid; i < NN * 3; i += 256) sx[i] = xyz[(long)b * NN * 3 + i];
  __syncthreads();
  const float qx = sx[n * 3 + 0], qy = sx[n * 3 + 1], qz = sx[n * 3 + 2];
  for (int m = tid; m < NN; m += 256) {
    float dx = qx - sx[m * 3 + 0];
    float dy = qy - sx[m * 3 + 1];
    float dz = qz - sx[m * 3 + 2];
    sd[m] = dx * dx + dy * dy + dz * dz;
  }
  __syncthreads();
  for (int it = 0; it < KNB; ++it) {
    float bv = 3.4e38f;
    int bi = NN;
    for (int m = tid; m < NN; m += 256) {
      float v = sd[m];
      if (v < bv) { bv = v; bi = m; }
    }
    rv[tid] = bv; ri[tid] = bi;
    __syncthreads();
    for (int s = 128; s > 0; s >>= 1) {
      if (tid < s) {
        if (rv[tid + s] < rv[tid] ||
            (rv[tid + s] == rv[tid] && ri[tid + s] < ri[tid])) {
          rv[tid] = rv[tid + s];
          ri[tid] = ri[tid + s];
        }
      }
      __syncthreads();
    }
    if (tid == 0) { knn[(long)p * KNB + it] = ri[0]; sd[ri[0]] = 3.4e38f; }
    __syncthreads();
  }
}

// ---------------------------------------------------------------------------
// Kernel 2: x = fc1(features); q,k,v = x @ w^T  (one block per point)
// ---------------------------------------------------------------------------
__global__ __launch_bounds__(128) void fc1qkv_kernel(
    const float* __restrict__ feat, const float* __restrict__ w1, const float* __restrict__ b1,
    const float* __restrict__ wq, const float* __restrict__ wk, const float* __restrict__ wv,
    float* __restrict__ x, float* __restrict__ q, float* __restrict__ k, float* __restrict__ v) {
  __shared__ float sf[DP];
  __shared__ float sx[DM];
  const int tid = threadIdx.x;
  const long p = blockIdx.x;
  if (tid < DP) sf[tid] = feat[p * DP + tid];
  __syncthreads();
  float acc = b1[tid];
#pragma unroll 8
  for (int d = 0; d < DP; ++d) acc += sf[d] * w1[tid * DP + d];
  sx[tid] = acc;
  x[p * DM + tid] = acc;
  __syncthreads();
  float aq = 0.f, ak = 0.f, av = 0.f;
#pragma unroll 4
  for (int i = 0; i < DM; ++i) {
    float xv = sx[i];
    aq += xv * wq[tid * DM + i];
    ak += xv * wk[tid * DM + i];
    av += xv * wv[tid * DM + i];
  }
  q[p * DM + tid] = aq;
  k[p * DM + tid] = ak;
  v[p * DM + tid] = av;
}

// ---------------------------------------------------------------------------
// Kernel 3: the WMMA attention core.
// One block = 4 points -> 144 rows = 9 M=16 tiles, 9 waves (288 threads).
// Per-tile pipeline: h1 (VALU) -> delta GEMM (WMMA) -> a0 -> gamma1 GEMM+ReLU
//                    -> gamma2 GEMM -> L1 norm -> attn out + res accumulation.
// ---------------------------------------------------------------------------
struct __align__(32) WaveScratch {
  _Float16 pos[16 * DM];   // 4 KB
  _Float16 a0[16 * DM];    // 4 KB
  _Float16 h2[16 * DM];    // 4 KB
  float    rel[16][4];
  int      gidx[16];
  int      ptx[16];
  float    rowsum[16];
};

__global__ __launch_bounds__(288, 1) void attn_kernel(
    const float* __restrict__ xyz, const int* __restrict__ knn,
    const float* __restrict__ qg, const float* __restrict__ kg, const float* __restrict__ vg,
    const float* __restrict__ dw1, const float* __restrict__ db1,
    const float* __restrict__ dw2, const float* __restrict__ db2,
    const float* __restrict__ gw1, const float* __restrict__ gb1,
    const float* __restrict__ gw2, const float* __restrict__ gb2,
    float* __restrict__ res_pre, float* __restrict__ attn_out) {
  __shared__ __align__(32) _Float16 sWD2[DM * DM];
  __shared__ __align__(32) _Float16 sWG1[DM * DM];
  __shared__ __align__(32) _Float16 sWG2[DM * DM];
  __shared__ float sD1[DM * 3];
  __shared__ float sDB1[DM], sDB2[DM], sGB1[DM], sGB2[DM];
  __shared__ float sQ[4 * DM];
  __shared__ float sRes[4 * DM];
  __shared__ WaveScratch sws[9];

  const int tid = threadIdx.x;
  const int p0 = blockIdx.x * 4;  // first global point of this block

  // ---- stage weights (f32 -> f16), biases, q rows; zero res accumulator ----
  for (int i = tid; i < DM * DM; i += 288) {
    sWD2[i] = (_Float16)dw2[i];
    sWG1[i] = (_Float16)gw1[i];
    sWG2[i] = (_Float16)gw2[i];
  }
  for (int i = tid; i < DM * 3; i += 288) sD1[i] = dw1[i];
  for (int i = tid; i < DM; i += 288) {
    sDB1[i] = db1[i]; sDB2[i] = db2[i]; sGB1[i] = gb1[i]; sGB2[i] = gb2[i];
  }
  for (int i = tid; i < 4 * DM; i += 288) {
    sQ[i] = qg[(long)(p0 + i / DM) * DM + (i % DM)];
    sRes[i] = 0.f;
  }

  const int wvid = tid >> 5;
  const int lane = tid & 31;
  const int hi = lane >> 4;   // half-wave select
  const int lm = lane & 15;   // row (A/D) or column (B/D) index
  WaveScratch* S = &sws[wvid];

  // ---- per-row metadata: row = wvid*16 + lane (lanes 0..15) ----
  if (lane < 16) {
    int rowg = wvid * 16 + lane;           // 0..143
    int pt = rowg / KNB;                   // point within block (0..3)
    int kn = rowg % KNB;                   // neighbor index
    int p = p0 + pt;
    int b = p / NN;
    int idx = knn[(long)p * KNB + kn];
    int gp = b * NN + idx;                 // gathered global point
    S->gidx[lane] = gp;
    S->ptx[lane] = pt;
    S->rel[lane][0] = xyz[(long)p * 3 + 0] - xyz[(long)gp * 3 + 0];
    S->rel[lane][1] = xyz[(long)p * 3 + 1] - xyz[(long)gp * 3 + 1];
    S->rel[lane][2] = xyz[(long)p * 3 + 2] - xyz[(long)gp * 3 + 2];
  }
  __syncthreads();

  // ---- Step A: h1 = relu(delta_w1 * rel + b1), built directly as A-frags ----
  v16h A[4];
  {
    const float r0 = S->rel[lm][0], r1 = S->rel[lm][1], r2 = S->rel[lm][2];
#pragma unroll
    for (int kc = 0; kc < 4; ++kc) {
      const int g0 = kc * 32 + hi * 8;
      const int g1 = kc * 32 + 16 + hi * 8;
      v16h a;
#pragma unroll
      for (int t = 0; t < 8; ++t) {
        int i0 = g0 + t, i1 = g1 + t;
        float h0 = sDB1[i0] + sD1[i0 * 3] * r0 + sD1[i0 * 3 + 1] * r1 + sD1[i0 * 3 + 2] * r2;
        float h1 = sDB1[i1] + sD1[i1 * 3] * r0 + sD1[i1 * 3 + 1] * r1 + sD1[i1 * 3 + 2] * r2;
        a[t]     = (_Float16)fmaxf(h0, 0.f);
        a[t + 8] = (_Float16)fmaxf(h1, 0.f);
      }
      A[kc] = a;
    }
  }

  // ---- Step B: pos = h1 @ delta_w2^T + b2  (8 x 4 WMMAs) ----
#pragma unroll
  for (int nt = 0; nt < 8; ++nt) {
    float bsc = sDB2[nt * 16 + lm];
    v8f acc;
#pragma unroll
    for (int j = 0; j < 8; ++j) acc[j] = bsc;
#pragma unroll
    for (int kc = 0; kc < 4; ++kc)
      acc = wmma32(A[kc], load_b_frag(sWD2, nt, kc, lm, hi), acc);
#pragma unroll
    for (int j = 0; j < 8; ++j)
      S->pos[(j + 8 * hi) * DM + nt * 16 + lm] = (_Float16)acc[j];
  }
  asm volatile("s_wait_dscnt 0" ::: "memory");

  // ---- Step C: a0 = q - k[knn] + pos (elementwise, gathered k) ----
#pragma unroll 4
  for (int e = 0; e < 64; ++e) {
    int flat = e * 32 + lane;
    int m = flat >> 7, c = flat & (DM - 1);
    float a = sQ[S->ptx[m] * DM + c] - kg[(long)S->gidx[m] * DM + c] + (float)S->pos[m * DM + c];
    S->a0[m * DM + c] = (_Float16)a;
  }
  asm volatile("s_wait_dscnt 0" ::: "memory");

  // ---- Step D: h2 = relu(a0 @ gamma_w1^T + b) ----
#pragma unroll
  for (int kc = 0; kc < 4; ++kc) A[kc] = load_a_frag(S->a0, kc, lm, hi);
#pragma unroll
  for (int nt = 0; nt < 8; ++nt) {
    float bsc = sGB1[nt * 16 + lm];
    v8f acc;
#pragma unroll
    for (int j = 0; j < 8; ++j) acc[j] = bsc;
#pragma unroll
    for (int kc = 0; kc < 4; ++kc)
      acc = wmma32(A[kc], load_b_frag(sWG1, nt, kc, lm, hi), acc);
#pragma unroll
    for (int j = 0; j < 8; ++j)
      S->h2[(j + 8 * hi) * DM + nt * 16 + lm] = (_Float16)fmaxf(acc[j], 0.f);
  }
  asm volatile("s_wait_dscnt 0" ::: "memory");

  // ---- Step E: attn_raw = h2 @ gamma_w2^T + b (accumulators stay in regs) ----
#pragma unroll
  for (int kc = 0; kc < 4; ++kc) A[kc] = load_a_frag(S->h2, kc, lm, hi);
  v8f att[8];
#pragma unroll
  for (int nt = 0; nt < 8; ++nt) {
    float bsc = sGB2[nt * 16 + lm];
    v8f acc;
#pragma unroll
    for (int j = 0; j < 8; ++j) acc[j] = bsc;
#pragma unroll
    for (int kc = 0; kc < 4; ++kc)
      acc = wmma32(A[kc], load_b_frag(sWG2, nt, kc, lm, hi), acc);
    att[nt] = acc;
  }

  // ---- Step F: L1-ish normalization per row via LDS f32 atomics ----
  if (lane < 16) S->rowsum[lane] = 0.f;
  asm volatile("s_wait_dscnt 0" ::: "memory");
  float ps[8];
#pragma unroll
  for (int j = 0; j < 8; ++j) {
    float s = 0.f;
#pragma unroll
    for (int nt = 0; nt < 8; ++nt) s += fabsf(att[nt][j]);
    ps[j] = s;
  }
#pragma unroll
  for (int j = 0; j < 8; ++j) atomicAdd(&S->rowsum[j + 8 * hi], ps[j]);
  asm volatile("s_wait_dscnt 0" ::: "memory");
  float inv[8];
#pragma unroll
  for (int j = 0; j < 8; ++j)
    inv[j] = 1.f / (S->rowsum[j + 8 * hi] + (float)DM * 1e-5f);
#pragma unroll
  for (int nt = 0; nt < 8; ++nt)
#pragma unroll
    for (int j = 0; j < 8; ++j) att[nt][j] *= inv[j];

  // ---- Step G: write attn, accumulate res += attn * (v[knn] + pos) ----
  int gidxm[8], ptm[8];
#pragma unroll
  for (int j = 0; j < 8; ++j) {
    gidxm[j] = S->gidx[j + 8 * hi];
    ptm[j]   = S->ptx[j + 8 * hi];
  }
  const long arowbase = ((long)p0 * KNB + wvid * 16) * DM;
#pragma unroll
  for (int nt = 0; nt < 8; ++nt) {
    const int c = nt * 16 + lm;
#pragma unroll
    for (int j = 0; j < 8; ++j) {
      const int m = j + 8 * hi;
      const float av = att[nt][j];
      attn_out[arowbase + (long)m * DM + c] = av;
      float contrib = av * (vg[(long)gidxm[j] * DM + c] + (float)S->pos[m * DM + c]);
      atomicAdd(&sRes[ptm[j] * DM + c], contrib);
    }
  }
  __syncthreads();
  for (int i = tid; i < 4 * DM; i += 288)
    res_pre[(long)(p0 + i / DM) * DM + (i % DM)] = sRes[i];
}

// ---------------------------------------------------------------------------
// MTLinear helper kernels
// ---------------------------------------------------------------------------
__global__ __launch_bounds__(128) void mean_kernel(const float* __restrict__ src,
                                                   float* __restrict__ mf) {
  const int b = blockIdx.x, tid = threadIdx.x;
  float s = 0.f;
  for (int n = 0; n < NN; ++n) s += src[((long)b * NN + n) * DM + tid];
  mf[b * DM + tid] = s * (1.f / (float)NN);
}

__global__ __launch_bounds__(128) void wcgen_kernel(const float* __restrict__ mf,
                                                    const float* __restrict__ cw,
                                                    const float* __restrict__ cb,
                                                    float* __restrict__ wc) {
  __shared__ float sm[DM];
  const int b = blockIdx.x >> 7;
  const int jb = (blockIdx.x & 127) * 128;
  const int tid = threadIdx.x;
  sm[tid] = mf[b * DM + tid];
  __syncthreads();
  const int j = jb + tid;
  float acc = cb[j];
#pragma unroll 4
  for (int c = 0; c < DM; ++c) acc += sm[c] * cw[(long)j * DM + c];
  wc[(long)b * DM * DM + j] = acc;
}

__global__ __launch_bounds__(128) void colsum_kernel(const float* __restrict__ wc,
                                                     float* __restrict__ cs) {
  const int b = blockIdx.x, i = threadIdx.x;
  float s = (float)DM * 1e-5f;
  for (int o = 0; o < DM; ++o) s += fabsf(wc[(long)b * DM * DM + o * DM + i]);
  cs[b * DM + i] = s;
}

// dst[p,o] = sum_c (src[p,c]/cs[b,c]) * wc[b,o,c] + addres[p,o]   (in-place safe)
__global__ __launch_bounds__(128) void bmm_kernel(const float* __restrict__ src,
                                                  const float* __restrict__ wc,
                                                  const float* __restrict__ cs,
                                                  const float* __restrict__ addres,
                                                  float* __restrict__ dst) {
  __shared__ float ss[DM];
  const long p = blockIdx.x;
  const int b = (int)(p / NN);
  const int tid = threadIdx.x;
  ss[tid] = src[p * DM + tid] / cs[b * DM + tid];
  __syncthreads();
  float acc = addres[p * DM + tid];
  const float* w = wc + (long)b * DM * DM + (long)tid * DM;
#pragma unroll 4
  for (int c = 0; c < DM; ++c) acc += ss[c] * w[c];
  dst[p * DM + tid] = acc;
}

// ---------------------------------------------------------------------------
// Host launcher
// ---------------------------------------------------------------------------
extern "C" void kernel_launch(void* const* d_in, const int* in_sizes, int n_in,
                              void* d_out, int out_size, void* d_ws, size_t ws_size,
                              hipStream_t stream) {
  (void)in_sizes; (void)n_in; (void)out_size; (void)ws_size;
  const float* features = (const float*)d_in[0];
  const float* xyz      = (const float*)d_in[1];
  const float* fc1_w    = (const float*)d_in[2];
  const float* fc1_b    = (const float*)d_in[3];
  const float* wq       = (const float*)d_in[4];
  const float* wk       = (const float*)d_in[5];
  const float* wv       = (const float*)d_in[6];
  const float* dw1      = (const float*)d_in[7];
  const float* db1      = (const float*)d_in[8];
  const float* dw2      = (const float*)d_in[9];
  const float* db2      = (const float*)d_in[10];
  const float* gw1      = (const float*)d_in[11];
  const float* gb1      = (const float*)d_in[12];
  const float* gw2      = (const float*)d_in[13];
  const float* gb2      = (const float*)d_in[14];
  const float* fc2_cw   = (const float*)d_in[15];
  const float* fc2_cb   = (const float*)d_in[16];
  const float* fc3_cw   = (const float*)d_in[17];
  const float* fc3_cb   = (const float*)d_in[18];

  const long PTS = (long)BB * NN;          // 8192
  const long PD  = PTS * DM;               // 1,048,576

  float* ws      = (float*)d_ws;
  float* x_ws    = ws;
  float* q_ws    = ws + PD;
  float* k_ws    = ws + 2 * PD;
  float* v_ws    = ws + 3 * PD;
  float* res_pre = ws + 4 * PD;
  int*   knn_ws  = (int*)(ws + 5 * PD);    // 294,912 ints
  float* mf_ws   = ws + 5 * PD + 294912;
  float* wc_ws   = mf_ws + 512;
  float* cs_ws   = wc_ws + BB * DM * DM;

  float* out      = (float*)d_out;
  float* attn_out = out + PD;              // (B,N,K,DM) after res

  knn_kernel<<<(int)PTS, 256, 0, stream>>>(xyz, knn_ws);
  fc1qkv_kernel<<<(int)PTS, 128, 0, stream>>>(features, fc1_w, fc1_b, wq, wk, wv,
                                              x_ws, q_ws, k_ws, v_ws);
  attn_kernel<<<(int)(PTS / 4), 288, 0, stream>>>(xyz, knn_ws, q_ws, k_ws, v_ws,
                                                  dw1, db1, dw2, db2,
                                                  gw1, gb1, gw2, gb2,
                                                  res_pre, attn_out);
  // res = MTLinear_fc2(res_pre) + x
  mean_kernel  <<<BB, 128, 0, stream>>>(res_pre, mf_ws);
  wcgen_kernel <<<BB * 128, 128, 0, stream>>>(mf_ws, fc2_cw, fc2_cb, wc_ws);
  colsum_kernel<<<BB, 128, 0, stream>>>(wc_ws, cs_ws);
  bmm_kernel   <<<(int)PTS, 128, 0, stream>>>(res_pre, wc_ws, cs_ws, x_ws, out);
  // res = MTLinear_fc3(res) + res   (in place on d_out)
  mean_kernel  <<<BB, 128, 0, stream>>>(out, mf_ws);
  wcgen_kernel <<<BB * 128, 128, 0, stream>>>(mf_ws, fc3_cw, fc3_cb, wc_ws);
  colsum_kernel<<<BB, 128, 0, stream>>>(wc_ws, cs_ws);
  bmm_kernel   <<<(int)PTS, 128, 0, stream>>>(out, wc_ws, cs_ws, out, out);
}
